// MultiHeadedAttention_43009802502482
// MI455X (gfx1250) — matrix operand order repaired and checked by reading, not compiled, not run on hardware
//
#include <hip/hip_runtime.h>
#include <stdint.h>

// ---------------------------------------------------------------------------
// MI455X (gfx1250) multi-head attention, bf16 WMMA path.
//   B=8, S=1024, D=1024, H=16, DK=64
// q/k/v = X@W+b (bf16 in ws, head-split layout), flash attention with
// post-softmax group_prob modulation, out = x@Wo+bo (fp32).
// Attention K/Q tiles use GLOBAL_LOAD_ASYNC_TO_LDS_B128 (ASYNCcnt) with
// double buffering.
// ---------------------------------------------------------------------------

typedef __attribute__((ext_vector_type(16))) __bf16 v16bf;
typedef __attribute__((ext_vector_type(8)))  float  v8f;

union BF16Frag { uint32_t u[8]; v16bf v; };

__device__ __forceinline__ uint16_t f2bf(float f) {
  union { float f; uint32_t u; } x; x.f = f;
  uint32_t u = x.u;
  uint32_t r = u + 0x7FFFu + ((u >> 16) & 1u);   // round-to-nearest-even
  return (uint16_t)(r >> 16);
}

__device__ __forceinline__ v8f v8zero() {
  v8f z;
#pragma unroll
  for (int i = 0; i < 8; ++i) z[i] = 0.0f;
  return z;
}

__device__ __forceinline__ v8f wmma_bf16(const BF16Frag& a, const BF16Frag& b, v8f c) {
  return __builtin_amdgcn_wmma_f32_16x16x32_bf16(
      /*neg_a=*/false, a.v, /*neg_b=*/false, b.v,
      /*c_mod=*/(short)0, c, /*reuse_a=*/false, /*reuse_b=*/false);
}

// Async copy of a 64x64 bf16 tile (row stride 72 u16 = 144B in LDS) from a
// row-major [*, 64] bf16 global tensor. 512 16-byte chunks, 128 threads.
// Counts 4 on ASYNCcnt per wave; completion is in-order.
__device__ __forceinline__ void async_tile_64x64(const uint16_t* __restrict__ gbase,
                                                 uint32_t lds_base, int tid) {
#pragma unroll
  for (int c4 = 0; c4 < 4; ++c4) {
    int i = tid + c4 * 128;
    int r = i >> 3, c = i & 7;                        // row, 16B chunk in row
    uint32_t lds_off = lds_base + (uint32_t)(r * 144 + c * 16);
    uint64_t gaddr = (uint64_t)(uintptr_t)(gbase + (size_t)r * 64 + c * 8);
    asm volatile("global_load_async_to_lds_b128 %0, %1, off"
                 :: "v"(lds_off), "v"(gaddr) : "memory");
  }
}

// ---------------------------------------------------------------------------
// Tiled GEMM: C[M,N] = A[M,K] @ W[K,N] + bias[N]
// Workgroup: 128 threads = 4 wave32; tile 64(M) x 64(N), BK = 32.
// ---------------------------------------------------------------------------
template <bool A_BF16, bool OUT_HEADSPLIT>
__global__ __launch_bounds__(128) void gemm64(
    const void* __restrict__ Aptr, const float* __restrict__ W,
    const float* __restrict__ bias, void* __restrict__ Cptr,
    int M, int Kd, int Nd) {
  __shared__ uint16_t a_lds[64][34];   // [row][k]   bf16, padded (stride 68B)
  __shared__ uint16_t w_lds[64][34];   // [n][k]     bf16 (W tile transposed)

  const int tid  = threadIdx.x;
  const int lane = tid & 31;
  const int wave = tid >> 5;
  const int n0 = blockIdx.x * 64;
  const int m0 = blockIdx.y * 64;

  v8f acc[4];
#pragma unroll
  for (int t = 0; t < 4; ++t) acc[t] = v8zero();

  for (int k0 = 0; k0 < Kd; k0 += 32) {
    if constexpr (A_BF16) {
      const uint16_t* A16 = (const uint16_t*)Aptr;
      for (int i = tid; i < 64 * 16; i += 128) {       // u32 (bf16 pair) copies
        int r = i >> 4, c = (i & 15) * 2;
        *(uint32_t*)&a_lds[r][c] =
            *(const uint32_t*)&A16[(size_t)(m0 + r) * Kd + k0 + c];
      }
    } else {
      const float* Af = (const float*)Aptr;
      for (int i = tid; i < 64 * 32; i += 128) {
        int r = i >> 5, c = i & 31;
        a_lds[r][c] = f2bf(Af[(size_t)(m0 + r) * Kd + k0 + c]);
      }
    }
    for (int i = tid; i < 64 * 32; i += 128) {          // W tile, transposed
      int kk = i >> 6, n = i & 63;
      w_lds[n][kk] = f2bf(W[(size_t)(k0 + kk) * Nd + n0 + n]);
    }
    if (k0 + 32 < Kd) {                                 // prefetch next W tile
      const float* nw = &W[(size_t)(k0 + 32 + (tid >> 2)) * Nd + n0 + (tid & 3) * 16];
      __builtin_prefetch(nw, 0, 0);
    }
    __syncthreads();

    // A fragment (16x32 bf16): lane L holds M=L&15; K split by lane half.
    BF16Frag a;
    {
      int row = wave * 16 + (lane & 15);
      int kb  = (lane >> 4) * 16;
#pragma unroll
      for (int vv = 0; vv < 4; ++vv) {
        a.u[vv]     = *(const uint32_t*)&a_lds[row][kb + 2 * vv];
        a.u[4 + vv] = *(const uint32_t*)&a_lds[row][8 + kb + 2 * vv];
      }
    }
#pragma unroll
    for (int t = 0; t < 4; ++t) {
      // B fragment (32x16): lane L holds column n=L&15, K half by lane>>4.
      BF16Frag b;
      int n   = t * 16 + (lane & 15);
      int kb2 = (lane >> 4) * 16;
#pragma unroll
      for (int vv = 0; vv < 8; ++vv)
        b.u[vv] = *(const uint32_t*)&w_lds[n][kb2 + 2 * vv];
      acc[t] = wmma_bf16(a, b, acc[t]);
    }
    __syncthreads();
  }

  // C layout: lane half selects M 0..7 / 8..15 (VGPR r), lane&15 = N.
#pragma unroll
  for (int t = 0; t < 4; ++t) {
#pragma unroll
    for (int r = 0; r < 8; ++r) {
      int m = m0 + wave * 16 + (lane >> 4) * 8 + r;
      int n = n0 + t * 16 + (lane & 15);
      float val = acc[t][r] + bias[n];
      if constexpr (OUT_HEADSPLIT) {
        int bb = m >> 10, s = m & 1023;     // M = B*S rows
        int h = n >> 6, dk = n & 63;        // N = H*DK cols
        ((uint16_t*)Cptr)[((((size_t)bb * 16 + h) * 1024) + s) * 64 + dk] =
            f2bf(val);
      } else {
        ((float*)Cptr)[(size_t)m * Nd + n] = val;
      }
    }
  }
}

// ---------------------------------------------------------------------------
// Flash attention per (b,h): scores=qk^T/8, mask|diag, online softmax,
// p = softmax * group_prob (Z sums exp only), x = p @ v.
// WG = 128 threads (4 waves); 64 q-rows per WG; stream 64-key blocks.
// K tiles double-buffered via global_load_async_to_lds_b128.
// ---------------------------------------------------------------------------
__global__ __launch_bounds__(128) void attn64(
    const uint16_t* __restrict__ q, const uint16_t* __restrict__ k,
    const uint16_t* __restrict__ v, const float* __restrict__ gprob,
    const int* __restrict__ mask, uint16_t* __restrict__ xout) {
  __shared__ uint16_t q_lds[64][72];      // [qrow][dk]   (async, 144B stride)
  __shared__ uint16_t k_lds[2][64][72];   // [krow][dk]   (async, dbl-buffered)
  __shared__ uint16_t vt_lds[64][66];     // [dk][krow]   (manual transpose)
  __shared__ uint16_t p_lds[4][16][66];   // per-wave p tile [qrow][kcol]

  const int tid  = threadIdx.x;
  const int lane = tid & 31;
  const int wave = tid >> 5;
  const int bh = blockIdx.y;              // 0..127
  const int b = bh >> 4, h = bh & 15;
  const int q0 = blockIdx.x * 64;
  const size_t head_base = (size_t)bh * 1024 * 64;
  const size_t bSS = (size_t)b * 1024 * 1024;

  const int rr_half = (lane >> 4) * 8;    // C-layout row offset for this lane
  const int nn = lane & 15;               // C-layout column for this lane
  const int qrow_lane = q0 + wave * 16 + rr_half;
  const float scale = 0.125f;             // 1/sqrt(64)

  // Async: Q tile, then K tile 0. (In-order completion; each = 4 ops/wave.)
  async_tile_64x64(&q[head_base + (size_t)q0 * 64],
                   (uint32_t)(uintptr_t)&q_lds[0][0], tid);
  async_tile_64x64(&k[head_base],
                   (uint32_t)(uintptr_t)&k_lds[0][0][0], tid);

  float rowM[8], rowZ[8];
  v8f acc[4];
#pragma unroll
  for (int r = 0; r < 8; ++r) { rowM[r] = -3.0e38f; rowZ[r] = 0.0f; }
#pragma unroll
  for (int t = 0; t < 4; ++t) acc[t] = v8zero();

  for (int j = 0; j < 16; ++j) {
    const int kb0 = j * 64;
    const int buf = j & 1;
    if (j + 1 < 16)                                     // async K tile j+1
      async_tile_64x64(&k[head_base + (size_t)(kb0 + 64) * 64],
                       (uint32_t)(uintptr_t)&k_lds[buf ^ 1][0][0], tid);
    for (int i = tid; i < 64 * 64; i += 128) {          // V tile, transposed
      int key = i >> 6, dk = i & 63;
      vt_lds[dk][key] = v[head_base + (size_t)(kb0 + key) * 64 + dk];
    }
    // Wait for Q + K[j] (older async ops) to land; K[j+1] may stay in flight.
    if (j + 1 < 16) asm volatile("s_wait_asynccnt 0x4" ::: "memory");
    else            asm volatile("s_wait_asynccnt 0x0" ::: "memory");
    __syncthreads();

    // ----- scores = q @ k^T : A=q (K=dk), B=k^T (column n = key row n) -----
    BF16Frag aq0, aq1;
    {
      int row = wave * 16 + (lane & 15);
      int kb  = (lane >> 4) * 16;         // (lane>>4)*16
#pragma unroll
      for (int vv = 0; vv < 4; ++vv) {
        aq0.u[vv]     = *(const uint32_t*)&q_lds[row][kb + 2 * vv];
        aq0.u[4 + vv] = *(const uint32_t*)&q_lds[row][8 + kb + 2 * vv];
        aq1.u[vv]     = *(const uint32_t*)&q_lds[row][32 + kb + 2 * vv];
        aq1.u[4 + vv] = *(const uint32_t*)&q_lds[row][40 + kb + 2 * vv];
      }
    }
    v8f sc[4];
    const int kb2 = (lane >> 4) * 16;
    // Two passes over the 4 N-tiles: dependent WMMA pairs are 4 apart,
    // letting independent WMMAs fill the bf16 RAW hazard window.
#pragma unroll
    for (int t = 0; t < 4; ++t) {
      BF16Frag b0;
      int krow = t * 16 + nn;
#pragma unroll
      for (int vv = 0; vv < 8; ++vv)
        b0.u[vv] = *(const uint32_t*)&k_lds[buf][krow][kb2 + 2 * vv];
      sc[t] = wmma_bf16(aq0, b0, v8zero());
    }
#pragma unroll
    for (int t = 0; t < 4; ++t) {
      BF16Frag b1;
      int krow = t * 16 + nn;
#pragma unroll
      for (int vv = 0; vv < 8; ++vv)
        b1.u[vv] = *(const uint32_t*)&k_lds[buf][krow][32 + kb2 + 2 * vv];
      sc[t] = wmma_bf16(aq1, b1, sc[t]);
    }

    // ----- mask + online softmax + group_prob fold-in -----
#pragma unroll
    for (int r = 0; r < 8; ++r) {
      const int mrow = qrow_lane + r;     // query row within [0,1024)
      float s4[4];
#pragma unroll
      for (int t = 0; t < 4; ++t) {
        int ncol = kb0 + t * 16 + nn;
        float s = sc[t][r] * scale;
        int allowed = mask[bSS + (size_t)mrow * 1024 + ncol] | (mrow == ncol);
        s4[t] = allowed ? s : -1.0e9f;
      }
      float lmax = fmaxf(fmaxf(s4[0], s4[1]), fmaxf(s4[2], s4[3]));
#pragma unroll
      for (int off = 1; off < 16; off <<= 1)
        lmax = fmaxf(lmax, __shfl_xor(lmax, off, 32));
      float Mnew = fmaxf(rowM[r], lmax);
      float corr = __expf(rowM[r] - Mnew);
      rowM[r] = Mnew;
      float lsum = 0.0f;
#pragma unroll
      for (int t = 0; t < 4; ++t) {
        int ncol = kb0 + t * 16 + nn;
        float e = __expf(s4[t] - Mnew);
        lsum += e;                                   // Z sums exp only
        float gp = gprob[bSS + (size_t)mrow * 1024 + ncol];
        p_lds[wave][rr_half + r][t * 16 + nn] = f2bf(e * gp);
      }
#pragma unroll
      for (int off = 1; off < 16; off <<= 1)
        lsum += __shfl_xor(lsum, off, 32);
      rowZ[r] = rowZ[r] * corr + lsum;
#pragma unroll
      for (int t = 0; t < 4; ++t) acc[t][r] *= corr;
    }
    __builtin_amdgcn_wave_barrier();      // p_lds stores before re-pack (DS in-order)

    // ----- x += p @ v : A=p (K=key), B=v (column n = dk) -----
    BF16Frag ap0, ap1;
    {
      int row = lane & 15;
      int kb  = (lane >> 4) * 16;
#pragma unroll
      for (int vv = 0; vv < 4; ++vv) {
        ap0.u[vv]     = *(const uint32_t*)&p_lds[wave][row][kb + 2 * vv];
        ap0.u[4 + vv] = *(const uint32_t*)&p_lds[wave][row][8 + kb + 2 * vv];
        ap1.u[vv]     = *(const uint32_t*)&p_lds[wave][row][32 + kb + 2 * vv];
        ap1.u[4 + vv] = *(const uint32_t*)&p_lds[wave][row][40 + kb + 2 * vv];
      }
    }
#pragma unroll
    for (int t = 0; t < 4; ++t) {
      BF16Frag b0;
      int dkcol = t * 16 + nn;
#pragma unroll
      for (int vv = 0; vv < 8; ++vv)
        b0.u[vv] = *(const uint32_t*)&vt_lds[dkcol][kb2 + 2 * vv];
      acc[t] = wmma_bf16(ap0, b0, acc[t]);
    }
#pragma unroll
    for (int t = 0; t < 4; ++t) {
      BF16Frag b1;
      int dkcol = t * 16 + nn;
#pragma unroll
      for (int vv = 0; vv < 8; ++vv)
        b1.u[vv] = *(const uint32_t*)&vt_lds[dkcol][32 + kb2 + 2 * vv];
      acc[t] = wmma_bf16(ap1, b1, acc[t]);
    }
    __syncthreads();
  }

  // Normalize by Z and store x as bf16 in [B,S,D] (heads re-interleaved).
#pragma unroll
  for (int t = 0; t < 4; ++t) {
#pragma unroll
    for (int r = 0; r < 8; ++r) {
      int mrow = qrow_lane + r;
      int dk = t * 16 + nn;
      float val = acc[t][r] / rowZ[r];
      xout[((size_t)b * 1024 + mrow) * 1024 + h * 64 + dk] = f2bf(val);
    }
  }
}

// ---------------------------------------------------------------------------
extern "C" void kernel_launch(void* const* d_in, const int* in_sizes, int n_in,
                              void* d_out, int out_size, void* d_ws,
                              size_t ws_size, hipStream_t stream) {
  (void)in_sizes; (void)n_in; (void)out_size; (void)ws_size;
  const float* query = (const float*)d_in[0];
  const float* key_  = (const float*)d_in[1];
  const float* value = (const float*)d_in[2];
  const float* gprob = (const float*)d_in[3];
  const int*   mask  = (const int*)d_in[4];
  const float* Wq = (const float*)d_in[5];
  const float* bq = (const float*)d_in[6];
  const float* Wk = (const float*)d_in[7];
  const float* bk = (const float*)d_in[8];
  const float* Wv = (const float*)d_in[9];
  const float* bv = (const float*)d_in[10];
  const float* Wo = (const float*)d_in[11];
  const float* bo = (const float*)d_in[12];
  float* out = (float*)d_out;

  const size_t elems = (size_t)8192 * 1024;   // B*S x D
  uint16_t* qws = (uint16_t*)d_ws;            // bf16 workspaces: 4 x 16 MiB
  uint16_t* kws = qws + elems;
  uint16_t* vws = kws + elems;
  uint16_t* xws = vws + elems;

  dim3 grid(16, 128);   // N/64 x M/64   (M = B*S = 8192, N = D = 1024)
  dim3 block(128);

  gemm64<false, true><<<grid, block, 0, stream>>>(query, Wq, bq, qws, 8192, 1024, 1024);
  gemm64<false, true><<<grid, block, 0, stream>>>(key_,  Wk, bk, kws, 8192, 1024, 1024);
  gemm64<false, true><<<grid, block, 0, stream>>>(value, Wv, bv, vws, 8192, 1024, 1024);

  attn64<<<dim3(16, 128), block, 0, stream>>>(qws, kws, vws, gprob, mask, xws);

  gemm64<true, false><<<grid, block, 0, stream>>>(xws, Wo, bo, out, 8192, 1024, 1024);
}
